// ShiftWindowMSA_7404523618523
// MI455X (gfx1250) — matrix operand order repaired and checked by reading, not compiled, not run on hardware
//
#include <hip/hip_runtime.h>
#include <hip/hip_bf16.h>

typedef __attribute__((ext_vector_type(16))) _Float16 v16h;
typedef __attribute__((ext_vector_type(8)))  _Float16 v8h;
typedef __attribute__((ext_vector_type(8)))  float    v8f;

// ---------------------------------------------------------------------------
// WMMA helpers (gfx1250: v_wmma_f32_16x16x32_f16, fp32 accumulate)
// ---------------------------------------------------------------------------
__device__ __forceinline__ v8f wmma_f16(v16h a, v16h b, v8f c) {
    return __builtin_amdgcn_wmma_f32_16x16x32_f16(
        /*neg_a=*/false, a, /*neg_b=*/false, b,
        /*c_mod=*/(short)0, c, /*reuse_a=*/false, /*reuse_b=*/false);
}

// A fragment: 16x32 f16 tile, row-major source with row stride lda (elements).
// Lane l: row = l&15, K = (l>>4)*16 + e (contig).
__device__ __forceinline__ v16h load_a_frag(const _Float16* A, int lda, int lane) {
    int row = lane & 15;
    int kb  = (lane >> 4) << 4;
    const _Float16* p = A + row * lda + kb;
    v8h lo = *(const v8h*)(p);
    v8h hi = *(const v8h*)(p + 8);
    v16h r;
#pragma unroll
    for (int e = 0; e < 8; ++e) { r[e] = lo[e]; r[e + 8] = hi[e]; }
    return r;
}

// B fragment: 32x16 f16 tile where B[k][n] is stored n-major (Bt[n][k]),
// row stride ldb (elements). Lane l: col = l&15, K = (l>>4)*16 + e (contig).
__device__ __forceinline__ v16h load_b_frag(const _Float16* Bt, int ldb, int lane) {
    int col = lane & 15;
    int kb  = (lane >> 4) << 4;
    return *(const v16h*)(Bt + col * ldb + kb);
}

// ---------------------------------------------------------------------------
// Constants:  B=16, H=W=64, C=512, NH=16, hd=32, WS=8, SS=4, N=64, nW=64
// ---------------------------------------------------------------------------
#define CDIM   512
#define NHEAD  16
#define HD     32
#define NTOK   64
#define NWIN   1024   /* B * nW = 16 * 64 */
#define MROWS  65536  /* NWIN * NTOK */

// region label boundaries for the shift mask: h<56 -> 0, h<60 -> 1, else 2
__device__ __forceinline__ int region1(int x) { return x < 56 ? 0 : (x < 60 ? 1 : 2); }
__device__ __forceinline__ int tok_label(int wh, int ww, int tok) {
    int h = (wh << 3) + (tok >> 3);
    int w = (ww << 3) + (tok & 7);
    return region1(h) * 3 + region1(w);
}

// ---------------------------------------------------------------------------
// Kernel 1: cyclic shift + window partition gather, fp32 -> f16
// ---------------------------------------------------------------------------
__global__ void __launch_bounds__(256)
gather_convert(const float* __restrict__ q, _Float16* __restrict__ xw) {
    size_t idx = (size_t)blockIdx.x * 256 + threadIdx.x;   // MROWS*CDIM elems
    int c    = (int)(idx & 511);
    size_t row = idx >> 9;
    int tok  = (int)(row & 63);
    int winl = (int)(row >> 6);
    int b  = winl >> 6;
    int w  = winl & 63;
    int wh = w >> 3, ww = w & 7;
    int i  = tok >> 3, j = tok & 7;
    int h  = ((wh << 3) + i + 4) & 63;        // roll(-SS): src = (pos+SS) mod H
    int wf = ((ww << 3) + j + 4) & 63;
    xw[idx] = (_Float16)q[((size_t)b * 4096 + h * 64 + wf) * CDIM + c];
}

// ---------------------------------------------------------------------------
// Kernel 2: weight conversion fp32 -> f16 (keep [n][k] layout)
// ---------------------------------------------------------------------------
__global__ void __launch_bounds__(256)
convert_weights(const float* __restrict__ qkv_w, const float* __restrict__ proj_w,
                _Float16* __restrict__ wq, _Float16* __restrict__ wp) {
    int idx = blockIdx.x * 256 + threadIdx.x;      // 786432 + 262144 elems
    if (idx < 786432) wq[idx] = (_Float16)qkv_w[idx];
    else              wp[idx - 786432] = (_Float16)proj_w[idx - 786432];
}

// ---------------------------------------------------------------------------
// Kernel 3: relative position bias  ->  bias[nh][i][j]  (fp32)
// ---------------------------------------------------------------------------
__global__ void __launch_bounds__(256)
build_bias(const float* __restrict__ table, const int* __restrict__ rel,
           float* __restrict__ bias) {
    int idx = blockIdx.x * 256 + threadIdx.x;      // NHEAD*4096 elems
    int nh = idx >> 12;
    int ij = idx & 4095;
    bias[idx] = table[rel[ij] * NHEAD + nh];
}

// ---------------------------------------------------------------------------
// Kernel 4: QKV GEMM  [65536 x 512] x [512 x 1536]^T(+b).
// 64x64 macro-tile per wave (4x4 WMMA tiles): 16 WMMA per 16 b128 loads per
// k-step -> 2x the arithmetic intensity of a 2x2 tiling. B fragments get a
// short live range (load one, use against all four A fragments).
// Epilogue scatters into Q (scaled), K ([win][head][tok][hd]) and
// V transposed ([win][head][hd][tok]).
// ---------------------------------------------------------------------------
__global__ void __launch_bounds__(256)
qkv_gemm(const _Float16* __restrict__ xw, const _Float16* __restrict__ wq,
         const float* __restrict__ qkv_b,
         _Float16* __restrict__ Q, _Float16* __restrict__ K,
         _Float16* __restrict__ V) {
    const int lane = threadIdx.x & 31;
    const int wid  = (blockIdx.x * 256 + threadIdx.x) >> 5;   // macro-tile id
    const int NT = 24;                       // 1536 / 64
    const int nt = wid % NT, mt = wid / NT;  // mt < 1024
    const int m0 = mt << 6, n0 = nt << 6;

    v8f acc[4][4] = {};
#pragma unroll 2
    for (int k0 = 0; k0 < CDIM; k0 += 32) {
        v16h a[4];
#pragma unroll
        for (int i = 0; i < 4; ++i)
            a[i] = load_a_frag(xw + (size_t)(m0 + (i << 4)) * CDIM + k0, CDIM, lane);
#pragma unroll
        for (int j = 0; j < 4; ++j) {
            v16h b = load_b_frag(wq + (size_t)(n0 + (j << 4)) * CDIM + k0, CDIM, lane);
#pragma unroll
            for (int i = 0; i < 4; ++i)
                acc[i][j] = wmma_f16(a[i], b, acc[i][j]);
        }
    }

    const float scale = 0.17677669529663687f;   // 32^-0.5
#pragma unroll
    for (int ii = 0; ii < 4; ++ii) {
#pragma unroll
        for (int jj = 0; jj < 4; ++jj) {
            int col   = n0 + (jj << 4) + (lane & 15);   // 0..1535
            int which = col >> 9;                       // 0=q 1=k 2=v
            int c     = col & 511;
            int head  = c >> 5, hdi = c & 31;
            float bv  = qkv_b[col];
#pragma unroll
            for (int r = 0; r < 8; ++r) {
                int row = m0 + (ii << 4) + r + ((lane >> 4) << 3);
                int win = row >> 6, tok = row & 63;
                float v = acc[ii][jj][r] + bv;
                size_t wh16 = (size_t)win * NHEAD + head;
                if (which == 0)
                    Q[(wh16 * NTOK + tok) * HD + hdi] = (_Float16)(v * scale);
                else if (which == 1)
                    K[(wh16 * NTOK + tok) * HD + hdi] = (_Float16)v;
                else
                    V[(wh16 * HD + hdi) * NTOK + tok] = (_Float16)v;
            }
        }
    }
}

// ---------------------------------------------------------------------------
// Kernel 5: attention, one wave per (window, head).
//   S = q k^T (16 WMMA) -> +bias +shift-mask -> LDS -> softmax -> P(f16)
//   O = P V   (16 WMMA) -> xa[win*64+tok][head*32+hdi]
// ---------------------------------------------------------------------------
#define ATT_WAVES 2
#define S_STRIDE  66
#define P_STRIDE  72

__global__ void __launch_bounds__(ATT_WAVES * 32)
attn_kernel(const _Float16* __restrict__ Q, const _Float16* __restrict__ K,
            const _Float16* __restrict__ V, const float* __restrict__ bias,
            _Float16* __restrict__ xa) {
    __shared__ __align__(32) float    Sls[ATT_WAVES][NTOK * S_STRIDE];
    __shared__ __align__(32) _Float16 Pls[ATT_WAVES][NTOK * P_STRIDE];

    const int lane  = threadIdx.x & 31;
    const int wslot = threadIdx.x >> 5;
    const int unit  = blockIdx.x * ATT_WAVES + wslot;   // 0..16383
    const int win   = unit >> 4;
    const int head  = unit & 15;

    float*    S = Sls[wslot];
    _Float16* P = Pls[wslot];

    const size_t wh16 = (size_t)win * NHEAD + head;
    const _Float16* qp = Q + wh16 * NTOK * HD;
    const _Float16* kp = K + wh16 * NTOK * HD;
    const _Float16* vp = V + wh16 * HD * NTOK;
    const float*    bp = bias + head * 4096;

    const int w  = win & 63;
    const int wh = w >> 3, ww = w & 7;

    // K^T as B fragments: Bt[n=tok][k=hd], ldb = HD
    v16h kf[4];
#pragma unroll
    for (int ct = 0; ct < 4; ++ct)
        kf[ct] = load_b_frag(kp + (size_t)(ct * 16) * HD, HD, lane);

    const int labj[4] = {
        tok_label(wh, ww, 0 * 16 + (lane & 15)),
        tok_label(wh, ww, 1 * 16 + (lane & 15)),
        tok_label(wh, ww, 2 * 16 + (lane & 15)),
        tok_label(wh, ww, 3 * 16 + (lane & 15))};

    // ---- S = q k^T, streamed row-tile by row-tile into LDS ----
#pragma unroll
    for (int mt = 0; mt < 4; ++mt) {
        v16h a = load_a_frag(qp + (size_t)(mt * 16) * HD, HD, lane);
        v8f s[4] = {};
#pragma unroll
        for (int ct = 0; ct < 4; ++ct) s[ct] = wmma_f16(a, kf[ct], s[ct]);

#pragma unroll
        for (int ct = 0; ct < 4; ++ct) {
            int j = (ct << 4) + (lane & 15);
#pragma unroll
            for (int r = 0; r < 8; ++r) {
                int i = (mt << 4) + r + ((lane >> 4) << 3);
                float val = s[ct][r] + bp[i * 64 + j];
                if (tok_label(wh, ww, i) != labj[ct]) val -= 100.0f;
                S[i * S_STRIDE + j] = val;
            }
        }
    }
    __syncthreads();

    // ---- row softmax: lane handles rows lane and lane+32 ----
#pragma unroll
    for (int rr = 0; rr < 2; ++rr) {
        int row = lane + (rr << 5);
        float m = -1e30f;
        for (int j = 0; j < 64; ++j) m = fmaxf(m, S[row * S_STRIDE + j]);
        float sum = 0.f;
        for (int j = 0; j < 64; ++j) {
            float e = __expf(S[row * S_STRIDE + j] - m);
            sum += e;
            S[row * S_STRIDE + j] = e;
        }
        float inv = 1.0f / sum;
        for (int j = 0; j < 64; ++j)
            P[row * P_STRIDE + j] = (_Float16)(S[row * S_STRIDE + j] * inv);
    }
    __syncthreads();

    // ---- O = P V : B frags from V^T ([hdi][tok], ldb = NTOK) ----
    v16h vf[2][2];
#pragma unroll
    for (int kk = 0; kk < 2; ++kk)
#pragma unroll
        for (int ct = 0; ct < 2; ++ct)
            vf[kk][ct] = load_b_frag(vp + (size_t)(ct * 16) * NTOK + kk * 32, NTOK, lane);

    _Float16* xout = xa + (size_t)win * NTOK * CDIM + head * HD;
#pragma unroll
    for (int mt = 0; mt < 4; ++mt) {
        v8f o[2] = {};
#pragma unroll
        for (int kk = 0; kk < 2; ++kk) {
            v16h pa = load_a_frag(P + mt * 16 * P_STRIDE + kk * 32, P_STRIDE, lane);
            o[0] = wmma_f16(pa, vf[kk][0], o[0]);
            o[1] = wmma_f16(pa, vf[kk][1], o[1]);
        }
#pragma unroll
        for (int ct = 0; ct < 2; ++ct) {
            int hdi = (ct << 4) + (lane & 15);
#pragma unroll
            for (int r = 0; r < 8; ++r) {
                int tok = (mt << 4) + r + ((lane >> 4) << 3);
                xout[(size_t)tok * CDIM + hdi] = (_Float16)o[ct][r];
            }
        }
    }
}

// ---------------------------------------------------------------------------
// Kernel 6: projection GEMM (64x64 per wave) + bias, fused window-reverse +
// reverse cyclic shift scatter to fp32 output.
// ---------------------------------------------------------------------------
__global__ void __launch_bounds__(256)
proj_gemm(const _Float16* __restrict__ xa, const _Float16* __restrict__ wp,
          const float* __restrict__ proj_b, float* __restrict__ out) {
    const int lane = threadIdx.x & 31;
    const int wid  = (blockIdx.x * 256 + threadIdx.x) >> 5;
    const int NT = 8;                        // 512 / 64
    const int nt = wid % NT, mt = wid / NT;  // mt < 1024
    const int m0 = mt << 6, n0 = nt << 6;

    v8f acc[4][4] = {};
#pragma unroll 2
    for (int k0 = 0; k0 < CDIM; k0 += 32) {
        v16h a[4];
#pragma unroll
        for (int i = 0; i < 4; ++i)
            a[i] = load_a_frag(xa + (size_t)(m0 + (i << 4)) * CDIM + k0, CDIM, lane);
#pragma unroll
        for (int j = 0; j < 4; ++j) {
            v16h b = load_b_frag(wp + (size_t)(n0 + (j << 4)) * CDIM + k0, CDIM, lane);
#pragma unroll
            for (int i = 0; i < 4; ++i)
                acc[i][j] = wmma_f16(a[i], b, acc[i][j]);
        }
    }

#pragma unroll
    for (int ii = 0; ii < 4; ++ii) {
#pragma unroll
        for (int jj = 0; jj < 4; ++jj) {
            int col  = n0 + (jj << 4) + (lane & 15);
            float bv = proj_b[col];
#pragma unroll
            for (int r = 0; r < 8; ++r) {
                int row = m0 + (ii << 4) + r + ((lane >> 4) << 3);
                int win = row >> 6, tok = row & 63;
                int b  = win >> 6;
                int w  = win & 63;
                int wh = w >> 3, ww = w & 7;
                int i  = tok >> 3, j = tok & 7;
                int h  = ((wh << 3) + i + 4) & 63;   // roll(+SS)
                int wf = ((ww << 3) + j + 4) & 63;
                out[((size_t)b * 4096 + h * 64 + wf) * CDIM + col] =
                    acc[ii][jj][r] + bv;
            }
        }
    }
}

// ---------------------------------------------------------------------------
// Launch
// ---------------------------------------------------------------------------
extern "C" void kernel_launch(void* const* d_in, const int* in_sizes, int n_in,
                              void* d_out, int out_size, void* d_ws, size_t ws_size,
                              hipStream_t stream) {
    const float* query      = (const float*)d_in[0];
    const float* qkv_w      = (const float*)d_in[1];
    const float* qkv_b      = (const float*)d_in[2];
    const float* proj_w     = (const float*)d_in[3];
    const float* proj_b     = (const float*)d_in[4];
    const float* bias_table = (const float*)d_in[5];
    const int*   rel_index  = (const int*)d_in[6];
    float* out = (float*)d_out;

    char*  ws  = (char*)d_ws;
    size_t off = 0;
    auto take = [&](size_t bytes) -> char* {
        char* p = ws + off;
        off = (off + bytes + 255) & ~(size_t)255;
        return p;
    };
    _Float16* xw   = (_Float16*)take((size_t)MROWS * CDIM * 2);      // 64 MB
    _Float16* wq   = (_Float16*)take((size_t)1536 * 512 * 2);        // 1.5 MB
    _Float16* wp   = (_Float16*)take((size_t)512 * 512 * 2);         // 0.5 MB
    _Float16* Qb   = (_Float16*)take((size_t)MROWS * CDIM * 2);      // 64 MB
    _Float16* Kb   = (_Float16*)take((size_t)MROWS * CDIM * 2);      // 64 MB
    _Float16* Vb   = (_Float16*)take((size_t)MROWS * CDIM * 2);      // 64 MB
    _Float16* xab  = (_Float16*)take((size_t)MROWS * CDIM * 2);      // 64 MB
    float*    bias = (float*)take((size_t)NHEAD * 4096 * 4);         // 256 KB

    // 1) shift + window gather, fp32 -> f16
    gather_convert<<<131072, 256, 0, stream>>>(query, xw);
    // 2) weights -> f16
    convert_weights<<<4096, 256, 0, stream>>>(qkv_w, proj_w, wq, wp);
    // 3) relative position bias table -> [nh][i][j]
    build_bias<<<256, 256, 0, stream>>>(bias_table, rel_index, bias);
    // 4) QKV GEMM (1024 * 24 macro-tiles of 64x64, 8 waves / block)
    qkv_gemm<<<3072, 256, 0, stream>>>(xw, wq, qkv_b, Qb, Kb, Vb);
    // 5) attention: 16384 (win,head) units, 2 waves / block
    attn_kernel<<<8192, ATT_WAVES * 32, 0, stream>>>(Qb, Kb, Vb, bias, xab);
    // 6) projection GEMM (1024 * 8 macro-tiles) + window reverse + shift
    proj_gemm<<<1024, 256, 0, stream>>>(xab, wp, proj_b, out);
}
